// RLIPv2_BiMultiHeadAttention_36558761623761
// MI455X (gfx1250) — matrix operand order repaired and checked by reading, not compiled, not run on hardware
//
#include <hip/hip_runtime.h>
#include <hip/hip_bf16.h>
#include <stdint.h>

#define B_   4
#define TV_  13600
#define TL_  256
#define VD_  256
#define LD_  768
#define E_   256
#define H_   8
#define HD_  32
#define CH_  160            // TV chunk for online softmax: 13600 = 85 * 160
#define SCALE_ 0.17677669529663687f   // 32^-0.5
#define CLAMP_ 50000.0f
#define NEG_  -9.0e15f

typedef __attribute__((ext_vector_type(16))) __bf16 v16bf;
typedef __attribute__((ext_vector_type(8)))  float  v8f;

union ABu { v16bf v; uint32_t u[8]; };

__device__ inline uint16_t f2bf(float f) {
  uint32_t u = __float_as_uint(f);
  u += 0x7fffu + ((u >> 16) & 1u);     // round-to-nearest-even
  return (uint16_t)(u >> 16);
}

__device__ inline v8f zero8() {
  v8f z = {0.f, 0.f, 0.f, 0.f, 0.f, 0.f, 0.f, 0.f};
  return z;
}

// A-operand (M=16 x K=32 bf16). lane = (m = lane&15, half = lane>>4).
// VGPR j<4 holds K pair (j + 4*half); VGPR j>=4 holds pair (j + 4 + 4*half).
__device__ inline void loadA(ABu& d, const uint16_t* rowk0) {
  const uint32_t* p = (const uint32_t*)rowk0;
  int h4 = ((threadIdx.x >> 4) & 1) * 4;
  d.u[0] = p[h4 + 0];  d.u[1] = p[h4 + 1];
  d.u[2] = p[h4 + 2];  d.u[3] = p[h4 + 3];
  d.u[4] = p[h4 + 8];  d.u[5] = p[h4 + 9];
  d.u[6] = p[h4 + 10]; d.u[7] = p[h4 + 11];
}

// B-operand (K=32 x N=16 bf16). lane = (n = lane&15, half).
// Holds B^T row n; VGPR j<4 holds K pair (j + 4*half); VGPR j>=4 holds pair (j + 4 + 4*half).
__device__ inline void loadB(ABu& d, const uint16_t* rowk0) {
  const uint32_t* p = (const uint32_t*)rowk0;
  int h8 = ((threadIdx.x >> 4) & 1) * 4;
#pragma unroll
  for (int j = 0; j < 8; ++j) d.u[j] = p[h8 + j + (j >= 4 ? 4 : 0)];
}

__device__ inline v8f wmma_bf16(const ABu& a, const ABu& b, v8f c) {
  return __builtin_amdgcn_wmma_f32_16x16x32_bf16(
      /*neg_a=*/false, a.v, /*neg_b=*/false, b.v,
      /*c_mod=*/(short)0, c, /*reuse_a=*/false, /*reuse_b=*/false);
}

// ---------------------------------------------------------------- converts
__global__ void cvt_kernel(const float* __restrict__ src, uint16_t* __restrict__ dst, int n) {
  int i = blockIdx.x * blockDim.x + threadIdx.x;
  if (i < n) dst[i] = f2bf(src[i]);
}

__global__ void addcvt_kernel(const float* __restrict__ a, const float* __restrict__ b,
                              uint16_t* __restrict__ dst, int n) {
  int i = blockIdx.x * blockDim.x + threadIdx.x;
  if (i < n) dst[i] = f2bf(a[i] + b[i]);
}

// LDS-tiled transpose: [Bz][T][E] -> [Bz][E][T], 32x32 tiles, coalesced both sides.
__global__ __launch_bounds__(256)
void transpose_tiled_kernel(const uint16_t* __restrict__ src, uint16_t* __restrict__ dst,
                            int T, int Ecols) {
  __shared__ uint16_t tile[32][33];
  int tx = threadIdx.x;            // 0..31
  int ty0 = threadIdx.y;           // 0..7
  int e0 = blockIdx.x * 32;
  int t0 = blockIdx.y * 32;
  int bz = blockIdx.z;
  const uint16_t* s = src + (size_t)bz * T * Ecols;
  uint16_t* d       = dst + (size_t)bz * Ecols * T;
#pragma unroll
  for (int j = 0; j < 4; ++j) {
    int ty = ty0 + j * 8;
    tile[ty][tx] = s[(size_t)(t0 + ty) * Ecols + e0 + tx];
  }
  __syncthreads();
#pragma unroll
  for (int j = 0; j < 4; ++j) {
    int ty = ty0 + j * 8;
    d[(size_t)(e0 + ty) * T + t0 + tx] = tile[tx][ty];
  }
}

// ---------------------------------------------------------------- GEMM
// out = (A[M,K](bf16) @ W[N,K](bf16)^T + bias) * scale
// One wave computes a 32x64 tile (8 WMMAs / K-step), software-pipelined loads.
// Block = 4 waves = 128x64 tile.
__global__ __launch_bounds__(128)
void gemm_bf16_kernel(const uint16_t* __restrict__ A, int lda,
                      const uint16_t* __restrict__ W, int ldb,
                      const float* __restrict__ bias,
                      int M, int N, int K, float scale,
                      uint16_t* __restrict__ outB, float* __restrict__ outF, int ldo) {
  int lane = threadIdx.x;
  int l16  = lane & 15;
  int half = lane >> 4;
  int row0 = (blockIdx.y * blockDim.y + threadIdx.y) * 32;
  int col0 = blockIdx.x * 64;
  if (row0 + 32 > M) return;

  const uint16_t* arow0 = A + (size_t)(row0 + l16) * lda;
  const uint16_t* arow1 = A + (size_t)(row0 + 16 + l16) * lda;
  const uint16_t* brow0 = W + (size_t)(col0 +  0 + l16) * ldb;
  const uint16_t* brow1 = W + (size_t)(col0 + 16 + l16) * ldb;
  const uint16_t* brow2 = W + (size_t)(col0 + 32 + l16) * ldb;
  const uint16_t* brow3 = W + (size_t)(col0 + 48 + l16) * ldb;

  v8f acc[2][4];
#pragma unroll
  for (int rr = 0; rr < 2; ++rr)
#pragma unroll
    for (int t = 0; t < 4; ++t) acc[rr][t] = zero8();

  ABu a0, a1, b0, b1, b2, b3;
  loadA(a0, arow0); loadA(a1, arow1);
  loadB(b0, brow0); loadB(b1, brow1); loadB(b2, brow2); loadB(b3, brow3);

  int k0 = 0;
  for (; k0 + 32 < K; k0 += 32) {
    int kn = k0 + 32;
    if (kn + 32 < K) {
      __builtin_prefetch(arow0 + kn + 32);
      __builtin_prefetch(arow1 + kn + 32);
    }
    ABu na0, na1, nb0, nb1, nb2, nb3;
    loadA(na0, arow0 + kn); loadA(na1, arow1 + kn);
    loadB(nb0, brow0 + kn); loadB(nb1, brow1 + kn);
    loadB(nb2, brow2 + kn); loadB(nb3, brow3 + kn);

    acc[0][0] = wmma_bf16(a0, b0, acc[0][0]);
    acc[1][0] = wmma_bf16(a1, b0, acc[1][0]);
    acc[0][1] = wmma_bf16(a0, b1, acc[0][1]);
    acc[1][1] = wmma_bf16(a1, b1, acc[1][1]);
    acc[0][2] = wmma_bf16(a0, b2, acc[0][2]);
    acc[1][2] = wmma_bf16(a1, b2, acc[1][2]);
    acc[0][3] = wmma_bf16(a0, b3, acc[0][3]);
    acc[1][3] = wmma_bf16(a1, b3, acc[1][3]);

    a0 = na0; a1 = na1; b0 = nb0; b1 = nb1; b2 = nb2; b3 = nb3;
  }
  acc[0][0] = wmma_bf16(a0, b0, acc[0][0]);
  acc[1][0] = wmma_bf16(a1, b0, acc[1][0]);
  acc[0][1] = wmma_bf16(a0, b1, acc[0][1]);
  acc[1][1] = wmma_bf16(a1, b1, acc[1][1]);
  acc[0][2] = wmma_bf16(a0, b2, acc[0][2]);
  acc[1][2] = wmma_bf16(a1, b2, acc[1][2]);
  acc[0][3] = wmma_bf16(a0, b3, acc[0][3]);
  acc[1][3] = wmma_bf16(a1, b3, acc[1][3]);

#pragma unroll
  for (int rr = 0; rr < 2; ++rr) {
#pragma unroll
    for (int t = 0; t < 4; ++t) {
      int col = col0 + t * 16 + l16;
      float bv = bias ? bias[col] : 0.f;
#pragma unroll
      for (int r = 0; r < 8; ++r) {
        int row = row0 + rr * 16 + r + half * 8;
        float val = (acc[rr][t][r] + bv) * scale;
        if (outB) outB[(size_t)row * ldo + col] = f2bf(val);
        if (outF) outF[(size_t)row * ldo + col] = val;
      }
    }
  }
}

// ---------------------------------------------------------------- attention: vision direction
// out_v_head[tile(16) x 32] = softmax_TL(clamp(q k^T) + ml) @ val_l.  One wave per 16-row tile.
__global__ __launch_bounds__(32)
void attn_v_kernel(const uint16_t* __restrict__ qb, const uint16_t* __restrict__ kb,
                   const uint16_t* __restrict__ vlT, const float* __restrict__ mask_l,
                   uint16_t* __restrict__ av) {
  __shared__ float    S[16][TL_];
  __shared__ uint16_t P[16][TL_];
  __shared__ float    inv[16];
  int lane = threadIdx.x, l16 = lane & 15, half = lane >> 4;
  int tile = blockIdx.x, h = blockIdx.y, b = blockIdx.z;

  ABu a; loadA(a, qb + ((size_t)b * TV_ + tile * 16 + l16) * E_ + h * HD_);

  for (int j = 0; j < TL_ / 16; ++j) {
    ABu kB; loadB(kB, kb + ((size_t)b * TL_ + j * 16 + l16) * E_ + h * HD_);
    v8f s8 = wmma_bf16(a, kB, zero8());
    int col = j * 16 + l16;
    float ml = mask_l[b * TL_ + col];
    ml = (ml == 0.f) ? NEG_ : ml;
#pragma unroll
    for (int r = 0; r < 8; ++r)
      S[r + half * 8][col] = fminf(fmaxf(s8[r], -CLAMP_), CLAMP_) + ml;
  }
  __syncthreads();
  if (lane < 16) {
    float mx = -3.0e38f;
    for (int c2 = 0; c2 < TL_; ++c2) mx = fmaxf(mx, S[lane][c2]);
    float sum = 0.f;
    for (int c2 = 0; c2 < TL_; ++c2) {
      float p = __expf(S[lane][c2] - mx);
      sum += p;
      P[lane][c2] = f2bf(p);
    }
    inv[lane] = 1.0f / sum;
  }
  __syncthreads();
  v8f o0 = zero8(), o1 = zero8();
  for (int kc = 0; kc < TL_ / 32; ++kc) {
    ABu pa; loadA(pa, &P[l16][kc * 32]);
    ABu b0; loadB(b0, vlT + ((size_t)b * E_ + h * HD_ + l16) * TL_ + kc * 32);
    ABu b1; loadB(b1, vlT + ((size_t)b * E_ + h * HD_ + 16 + l16) * TL_ + kc * 32);
    o0 = wmma_bf16(pa, b0, o0);
    o1 = wmma_bf16(pa, b1, o1);
  }
#pragma unroll
  for (int r = 0; r < 8; ++r) {
    int m = r + half * 8;
    size_t base = ((size_t)b * TV_ + tile * 16 + m) * E_ + h * HD_;
    av[base + l16]      = f2bf(o0[r] * inv[m]);
    av[base + 16 + l16] = f2bf(o1[r] * inv[m]);
  }
}

// ---------------------------------------------------------------- attention: language direction
// out_l_head = softmax_TV(clamp(k q^T) + mv) @ val_v, flash-style over TV in chunks of CH_.
__global__ __launch_bounds__(32)
void attn_l_kernel(const uint16_t* __restrict__ qb, const uint16_t* __restrict__ kb,
                   const uint16_t* __restrict__ vvT, const float* __restrict__ mask_v,
                   uint16_t* __restrict__ al) {
  __shared__ float    S[16][CH_];
  __shared__ uint16_t P[16][CH_];
  __shared__ float    scaleArr[16];
  __shared__ float    invArr[16];
  int lane = threadIdx.x, l16 = lane & 15, half = lane >> 4;
  int tile = blockIdx.x, h = blockIdx.y, b = blockIdx.z;

  ABu a; loadA(a, kb + ((size_t)b * TL_ + tile * 16 + l16) * E_ + h * HD_);

  float row_m = -3.0e38f, row_s = 0.f;   // valid in lanes < 16
  v8f o0 = zero8(), o1 = zero8();

  for (int c = 0; c < TV_ / CH_; ++c) {
    int tv0 = c * CH_;
    for (int j = 0; j < CH_ / 16; ++j) {
      ABu qB; loadB(qB, qb + ((size_t)b * TV_ + tv0 + j * 16 + l16) * E_ + h * HD_);
      v8f s8 = wmma_bf16(a, qB, zero8());
      int col = j * 16 + l16;
      float mv = mask_v[(size_t)b * TV_ + tv0 + col];
      mv = (mv == 0.f) ? NEG_ : mv;
#pragma unroll
      for (int r = 0; r < 8; ++r)
        S[r + half * 8][col] = fminf(fmaxf(s8[r], -CLAMP_), CLAMP_) + mv;
    }
    __syncthreads();
    if (lane < 16) {
      float mx = row_m;
      for (int c2 = 0; c2 < CH_; ++c2) mx = fmaxf(mx, S[lane][c2]);
      float sc  = __expf(row_m - mx);
      float sum = row_s * sc;
      for (int c2 = 0; c2 < CH_; ++c2) {
        float p = __expf(S[lane][c2] - mx);
        sum += p;
        P[lane][c2] = f2bf(p);
      }
      row_m = mx; row_s = sum;
      scaleArr[lane] = sc;
    }
    __syncthreads();
#pragma unroll
    for (int r = 0; r < 8; ++r) {
      float sc = scaleArr[r + half * 8];
      o0[r] *= sc; o1[r] *= sc;
    }
    for (int kc = 0; kc < CH_ / 32; ++kc) {
      ABu pa; loadA(pa, &P[l16][kc * 32]);
      ABu b0; loadB(b0, vvT + ((size_t)b * E_ + h * HD_ + l16) * TV_ + tv0 + kc * 32);
      ABu b1; loadB(b1, vvT + ((size_t)b * E_ + h * HD_ + 16 + l16) * TV_ + tv0 + kc * 32);
      o0 = wmma_bf16(pa, b0, o0);
      o1 = wmma_bf16(pa, b1, o1);
    }
    __syncthreads();
  }
  if (lane < 16) invArr[lane] = 1.0f / row_s;
  __syncthreads();
#pragma unroll
  for (int r = 0; r < 8; ++r) {
    int m = r + half * 8;
    size_t base = ((size_t)b * TL_ + tile * 16 + m) * E_ + h * HD_;
    al[base + l16]      = f2bf(o0[r] * invArr[m]);
    al[base + 16 + l16] = f2bf(o1[r] * invArr[m]);
  }
}

// ---------------------------------------------------------------- host
extern "C" void kernel_launch(void* const* d_in, const int* in_sizes, int n_in,
                              void* d_out, int out_size, void* d_ws, size_t ws_size,
                              hipStream_t stream) {
  (void)in_sizes; (void)n_in; (void)out_size; (void)ws_size;
  const float* v      = (const float*)d_in[0];
  const float* l      = (const float*)d_in[1];
  const float* v_pos  = (const float*)d_in[2];
  const float* mask_v = (const float*)d_in[3];
  const float* mask_l = (const float*)d_in[4];
  const float* Wq  = (const float*)d_in[5];  const float* bq  = (const float*)d_in[6];
  const float* Wk  = (const float*)d_in[7];  const float* bk  = (const float*)d_in[8];
  const float* Wvv = (const float*)d_in[9];  const float* bvv = (const float*)d_in[10];
  const float* Wvl = (const float*)d_in[11]; const float* bvl = (const float*)d_in[12];
  const float* Wov = (const float*)d_in[13]; const float* bov = (const float*)d_in[14];
  const float* Wol = (const float*)d_in[15]; const float* bol = (const float*)d_in[16];

  char* ws = (char*)d_ws;
  size_t cur = 0;
  auto carve = [&](size_t elems) -> uint16_t* {
    uint16_t* p = (uint16_t*)(ws + cur);
    cur += ((elems * 2 + 255) & ~(size_t)255);
    return p;
  };

  uint16_t* xq   = carve((size_t)B_ * TV_ * VD_);   // bf16(v + v_pos)
  uint16_t* xv   = carve((size_t)B_ * TV_ * VD_);   // bf16(v)
  uint16_t* xl   = carve((size_t)B_ * TL_ * LD_);   // bf16(l)
  uint16_t* qbuf = carve((size_t)B_ * TV_ * E_);
  uint16_t* kbuf = carve((size_t)B_ * TL_ * E_);
  uint16_t* vvb  = carve((size_t)B_ * TV_ * E_);
  uint16_t* vlb  = carve((size_t)B_ * TL_ * E_);
  uint16_t* vvT  = carve((size_t)B_ * E_ * TV_);
  uint16_t* vlT  = carve((size_t)B_ * E_ * TL_);
  uint16_t* avb  = carve((size_t)B_ * TV_ * E_);    // attention out (vision), bf16
  uint16_t* alb  = carve((size_t)B_ * TL_ * E_);    // attention out (language), bf16
  uint16_t* wqb  = carve((size_t)E_ * VD_);
  uint16_t* wkb  = carve((size_t)E_ * LD_);
  uint16_t* wvvb = carve((size_t)E_ * VD_);
  uint16_t* wvlb = carve((size_t)E_ * LD_);
  uint16_t* wovb = carve((size_t)VD_ * E_);
  uint16_t* wolb = carve((size_t)LD_ * E_);

  // --- converts
  int nVV = B_ * TV_ * VD_;
  int nL  = B_ * TL_ * LD_;
  addcvt_kernel<<<(nVV + 255) / 256, 256, 0, stream>>>(v, v_pos, xq, nVV);
  cvt_kernel<<<(nVV + 255) / 256, 256, 0, stream>>>(v, xv, nVV);
  cvt_kernel<<<(nL  + 255) / 256, 256, 0, stream>>>(l, xl, nL);
  cvt_kernel<<<(E_ * VD_ + 255) / 256, 256, 0, stream>>>(Wq,  wqb,  E_ * VD_);
  cvt_kernel<<<(E_ * LD_ + 255) / 256, 256, 0, stream>>>(Wk,  wkb,  E_ * LD_);
  cvt_kernel<<<(E_ * VD_ + 255) / 256, 256, 0, stream>>>(Wvv, wvvb, E_ * VD_);
  cvt_kernel<<<(E_ * LD_ + 255) / 256, 256, 0, stream>>>(Wvl, wvlb, E_ * LD_);
  cvt_kernel<<<(VD_ * E_ + 255) / 256, 256, 0, stream>>>(Wov, wovb, VD_ * E_);
  cvt_kernel<<<(LD_ * E_ + 255) / 256, 256, 0, stream>>>(Wol, wolb, LD_ * E_);

  // --- projections (wave tile 32x64, block tile 128x64)
  dim3 gb(32, 4);
  gemm_bf16_kernel<<<dim3(E_ / 64, (B_ * TV_) / 128), gb, 0, stream>>>(
      xq, VD_, wqb, VD_, bq, B_ * TV_, E_, VD_, SCALE_, qbuf, nullptr, E_);
  gemm_bf16_kernel<<<dim3(E_ / 64, (B_ * TL_) / 128), gb, 0, stream>>>(
      xl, LD_, wkb, LD_, bk, B_ * TL_, E_, LD_, 1.0f, kbuf, nullptr, E_);
  gemm_bf16_kernel<<<dim3(E_ / 64, (B_ * TV_) / 128), gb, 0, stream>>>(
      xv, VD_, wvvb, VD_, bvv, B_ * TV_, E_, VD_, 1.0f, vvb, nullptr, E_);
  gemm_bf16_kernel<<<dim3(E_ / 64, (B_ * TL_) / 128), gb, 0, stream>>>(
      xl, LD_, wvlb, LD_, bvl, B_ * TL_, E_, LD_, 1.0f, vlb, nullptr, E_);

  // --- transposes for PV B-operands (LDS-tiled, coalesced)
  transpose_tiled_kernel<<<dim3(E_ / 32, TV_ / 32, B_), dim3(32, 8), 0, stream>>>(
      vvb, vvT, TV_, E_);
  transpose_tiled_kernel<<<dim3(E_ / 32, TL_ / 32, B_), dim3(32, 8), 0, stream>>>(
      vlb, vlT, TL_, E_);

  // --- attention
  attn_v_kernel<<<dim3(TV_ / 16, H_, B_), 32, 0, stream>>>(qbuf, kbuf, vlT, mask_l, avb);
  attn_l_kernel<<<dim3(TL_ / 16, H_, B_), 32, 0, stream>>>(qbuf, kbuf, vvT, mask_v, alb);

  // --- output projections (fp32 to d_out)
  float* out_v = (float*)d_out;
  float* out_l = out_v + (size_t)B_ * TV_ * VD_;
  gemm_bf16_kernel<<<dim3(VD_ / 64, (B_ * TV_) / 128), gb, 0, stream>>>(
      avb, E_, wovb, E_, bov, B_ * TV_, VD_, E_, 1.0f, nullptr, out_v, VD_);
  gemm_bf16_kernel<<<dim3(LD_ / 64, (B_ * TL_) / 128), gb, 0, stream>>>(
      alb, E_, wolb, E_, bol, B_ * TL_, LD_, E_, 1.0f, nullptr, out_l, LD_);
}
